// MLPEncoder_42941083025725
// MI455X (gfx1250) — matrix operand order repaired and checked by reading, not compiled, not run on hardware
//
#include <hip/hip_runtime.h>

// ---------------------------------------------------------------------------
// MLPEncoder pipeline for MI455X (gfx1250, wave32):
//   h1 = BN(x @ W1 + b1)          bf16 WMMA GEMM (2-col register blocking)
//   h2 = BN(h1 @ W2 + b2)         bf16 WMMA GEMM
//   r  = capsule routing (k=8, layers=2, iters=3, m=20)
//        z tiles staged in LDS via GLOBAL_LOAD_ASYNC_TO_LDS_B128 (ASYNCcnt)
//   out= GCN agg (sym-norm adjacency, 1M edges + self loops)  f32 atomics
// ---------------------------------------------------------------------------

#define NODES_IN   500
#define KPAD       512
#define HID        512
#define OUTD       256
#define NCAPS      8
#define CAPD       32      // 256 / 8
#define NNBR       20

typedef __attribute__((ext_vector_type(16))) __bf16 v16bf;
typedef __attribute__((ext_vector_type(8)))  float  v8f;

union Frag32B { uint4 q[2]; v16bf v; };

__device__ __forceinline__ float wred(float v) {
  #pragma unroll
  for (int off = 16; off > 0; off >>= 1) v += __shfl_xor(v, off, 32);
  return v;
}

// --------------------- async global->LDS copy helpers ----------------------
// 16 bytes per lane per call; tracked by ASYNCcnt on gfx1250.
// Builtin prototype (from clang diagnostic): typed int4 pointers,
// global (AS1) source, LDS (AS3) destination, imm offset, imm cpol.

typedef int v4i_ __attribute__((vector_size(16)));
typedef __attribute__((address_space(1))) v4i_* gv4p;
typedef __attribute__((address_space(3))) v4i_* lv4p;

__device__ __forceinline__ void async_copy16(const float* gsrc, float* ldst) {
#if defined(__gfx1250__) && __has_builtin(__builtin_amdgcn_global_load_async_to_lds_b128)
  __builtin_amdgcn_global_load_async_to_lds_b128((gv4p)gsrc, (lv4p)ldst, 0, 0);
#else
  *(float4*)ldst = *(const float4*)gsrc;
#endif
}

__device__ __forceinline__ void async_wait0() {
#if defined(__gfx1250__) && __has_builtin(__builtin_amdgcn_s_wait_asynccnt)
  __builtin_amdgcn_s_wait_asynccnt(0);
#endif
}

// ------------------------- precision conversion ---------------------------

__global__ void conv_x_kernel(const float* __restrict__ x, __bf16* __restrict__ xb, int n) {
  int idx = blockIdx.x * blockDim.x + threadIdx.x;
  if (idx >= n * KPAD) return;
  int r = idx >> 9, k = idx & (KPAD - 1);
  xb[idx] = (k < NODES_IN) ? (__bf16)x[r * NODES_IN + k] : (__bf16)0.f;
}

// W1 [500,512] -> column-major padded Wt [512 cols][512 k]
__global__ void conv_w1_kernel(const float* __restrict__ W, __bf16* __restrict__ Wt) {
  int idx = blockIdx.x * blockDim.x + threadIdx.x;
  if (idx >= HID * KPAD) return;
  int ncol = idx >> 9, k = idx & (KPAD - 1);
  Wt[idx] = (k < NODES_IN) ? (__bf16)W[k * HID + ncol] : (__bf16)0.f;
}

// W2 [512,256] -> column-major Wt [256 cols][512 k]
__global__ void conv_w2_kernel(const float* __restrict__ W, __bf16* __restrict__ Wt) {
  int idx = blockIdx.x * blockDim.x + threadIdx.x;
  if (idx >= OUTD * KPAD) return;
  int ncol = idx >> 9, k = idx & (KPAD - 1);
  Wt[idx] = (__bf16)W[k * OUTD + ncol];
}

// ------------------------------- WMMA GEMM --------------------------------
// A: [M x 512] bf16 row-major; Bt: [CN cols x 512] bf16 (column-major B);
// C: [M x CN] f32.  One 16x32 tile per wave (A fragment reused for 2 WMMAs);
// 8 waves/block -> 256 output columns per block.
__global__ void __launch_bounds__(256)
gemm_bf16_kernel(const __bf16* __restrict__ A, const __bf16* __restrict__ Bt,
                 const float* __restrict__ bias, float* __restrict__ C, int CN) {
  const int wave = threadIdx.x >> 5;
  const int lane = threadIdx.x & 31;
  const int hb   = lane >> 4;          // half-wave select (K phase)
  const int l16  = lane & 15;
  const int tile_n = blockIdx.x * 256 + wave * 32;
  const int tile_m = blockIdx.y * 16;

  const __bf16* Arow  = A  + (size_t)(tile_m + l16) * KPAD;
  const __bf16* Bcol0 = Bt + (size_t)(tile_n + l16) * KPAD;
  const __bf16* Bcol1 = Bcol0 + (size_t)16 * KPAD;

  v8f acc0 = {}, acc1 = {};
  #pragma unroll 2
  for (int kb = 0; kb < KPAD; kb += 32) {
    Frag32B fa, fb0, fb1;
    // A fragment, contiguous k order: lanes 0-15 K=0..15, lanes 16-31 K=16..31 (matches B)
    fa.q[0] = *(const uint4*)(Arow + kb + hb * 16);
    fa.q[1] = *(const uint4*)(Arow + kb + hb * 16 + 8);
    // B fragment, contiguous k order: lanes 0-15 K=0..15, lanes 16-31 K=16..31 (col-major source)
    const uint4* bp0 = (const uint4*)(Bcol0 + kb + hb * 16);
    fb0.q[0] = bp0[0];
    fb0.q[1] = bp0[1];
    const uint4* bp1 = (const uint4*)(Bcol1 + kb + hb * 16);
    fb1.q[0] = bp1[0];
    fb1.q[1] = bp1[1];
    __builtin_prefetch(Arow + kb + 128, 0, 0);
    acc0 = __builtin_amdgcn_wmma_f32_16x16x32_bf16(false, fa.v, false, fb0.v,
                                                   (short)0, acc0, false, false);
    acc1 = __builtin_amdgcn_wmma_f32_16x16x32_bf16(false, fa.v, false, fb1.v,
                                                   (short)0, acc1, false, false);
  }
  const float bv0 = bias[tile_n + l16];
  const float bv1 = bias[tile_n + 16 + l16];
  float* Crow = C + (size_t)(tile_m + hb * 8) * CN + (tile_n + l16);
  #pragma unroll
  for (int i = 0; i < 8; ++i) {
    Crow[(size_t)i * CN]      = acc0[i] + bv0;
    Crow[(size_t)i * CN + 16] = acc1[i] + bv1;
  }
}

// ------------------------------- BatchNorm --------------------------------

__global__ void bnstats_kernel(const float* __restrict__ h, int C, int rows,
                               float* __restrict__ mean, float* __restrict__ rstd) {
  const int col = blockIdx.x;
  float s = 0.f, s2 = 0.f;
  for (int r = threadIdx.x; r < rows; r += blockDim.x) {
    float v = h[(size_t)r * C + col];
    s += v; s2 += v * v;
  }
  __shared__ float sh1[256], sh2[256];
  int t = threadIdx.x;
  sh1[t] = s; sh2[t] = s2; __syncthreads();
  for (int st = 128; st > 0; st >>= 1) {
    if (t < st) { sh1[t] += sh1[t + st]; sh2[t] += sh2[t + st]; }
    __syncthreads();
  }
  if (t == 0) {
    float mu  = sh1[0] / rows;
    float var = sh2[0] / rows - mu * mu;
    mean[col] = mu;
    rstd[col] = rsqrtf(var + 1e-5f);
  }
}

__global__ void bnapply_bf16_kernel(const float* __restrict__ h,
                                    const float* __restrict__ mean, const float* __restrict__ rstd,
                                    const float* __restrict__ gamma, const float* __restrict__ beta,
                                    __bf16* __restrict__ out, int total, int cmask) {
  int idx = blockIdx.x * blockDim.x + threadIdx.x;
  if (idx >= total) return;
  int c = idx & cmask;
  out[idx] = (__bf16)(gamma[c] * (h[idx] - mean[c]) * rstd[c] + beta[c]);
}

__global__ void bnapply_f32_kernel(const float* __restrict__ h,
                                   const float* __restrict__ mean, const float* __restrict__ rstd,
                                   const float* __restrict__ gamma, const float* __restrict__ beta,
                                   float* __restrict__ out, int total, int cmask) {
  int idx = blockIdx.x * blockDim.x + threadIdx.x;
  if (idx >= total) return;
  int c = idx & cmask;
  out[idx] = gamma[c] * (h[idx] - mean[c]) * rstd[c] + beta[c];
}

// ---------------------------- capsule routing -----------------------------
// Per-capsule l2norm; one wave handles one node (lane = capsule dim).
__global__ void capsnorm_kernel(const float* __restrict__ in, float* __restrict__ out, int n) {
  int node = (blockIdx.x * blockDim.x + threadIdx.x) >> 5;
  int lane = threadIdx.x & 31;
  if (node >= n) return;
  const float* src = in  + (size_t)node * 256;
  float*       dst = out + (size_t)node * 256;
  #pragma unroll
  for (int c = 0; c < NCAPS; ++c) {
    float v  = src[c * CAPD + lane];
    float ss = wred(v * v);
    dst[c * CAPD + lane] = v / fmaxf(sqrtf(ss), 1e-12f);
  }
}

// One wave per node; z tile (20x256 f32 = 20KB) staged in LDS via async
// global->LDS b128 transfers, reused across 3 routing iterations.
// 2 waves/block -> 40KB dynamic LDS (8 blocks/WGP within 320KB).
__global__ void route_kernel(const float* __restrict__ xn, const int* __restrict__ nbr,
                             float* __restrict__ out, int n) {
  extern __shared__ float zsh[];
  const int wslot = threadIdx.x >> 5;
  const int lane  = threadIdx.x & 31;
  const int node  = blockIdx.x * 2 + wslot;
  if (node >= n) return;
  float* z = zsh + wslot * (NNBR * 256);

  // stage z = xn[neighbors[node]]: 2 x b128 per lane per neighbor row
  for (int j = 0; j < NNBR; ++j) {
    int g = nbr[node * NNBR + j];
    const float* src = xn + (size_t)g * 256;
    float* dst = z + j * 256;
    async_copy16(src + lane * 4,       dst + lane * 4);
    async_copy16(src + 128 + lane * 4, dst + 128 + lane * 4);
  }

  float xc[NCAPS], u[NCAPS];
  #pragma unroll
  for (int c = 0; c < NCAPS; ++c) xc[c] = xn[(size_t)node * 256 + c * CAPD + lane];

  async_wait0();   // all async LDS writes for this wave's region complete

  // iter 0: p = 1/k (softmax of zeros)
  #pragma unroll
  for (int c = 0; c < NCAPS; ++c) {
    float s = 0.f;
    for (int j = 0; j < NNBR; ++j) s += z[j * 256 + c * CAPD + lane];
    u[c] = 0.125f * s + xc[c];
  }
  #pragma unroll
  for (int c = 0; c < NCAPS; ++c) {
    float ss = wred(u[c] * u[c]);
    u[c] /= fmaxf(sqrtf(ss), 1e-12f);
  }

  // iters 1..2: agreement logits -> softmax over capsules -> weighted agg
  for (int it = 1; it < 3; ++it) {
    float acc[NCAPS];
    #pragma unroll
    for (int c = 0; c < NCAPS; ++c) acc[c] = 0.f;
    for (int j = 0; j < NNBR; ++j) {
      float b[NCAPS];
      #pragma unroll
      for (int c = 0; c < NCAPS; ++c)
        b[c] = wred(z[j * 256 + c * CAPD + lane] * u[c]);   // replicated to all lanes
      float mx = b[0];
      #pragma unroll
      for (int c = 1; c < NCAPS; ++c) mx = fmaxf(mx, b[c]);
      float se = 0.f;
      #pragma unroll
      for (int c = 0; c < NCAPS; ++c) { b[c] = __expf(b[c] - mx); se += b[c]; }
      float inv = 1.f / se;
      #pragma unroll
      for (int c = 0; c < NCAPS; ++c)
        acc[c] += z[j * 256 + c * CAPD + lane] * (b[c] * inv);
    }
    #pragma unroll
    for (int c = 0; c < NCAPS; ++c) u[c] = acc[c] + xc[c];
    if (it < 2) {
      #pragma unroll
      for (int c = 0; c < NCAPS; ++c) {
        float ss = wred(u[c] * u[c]);
        u[c] /= fmaxf(sqrtf(ss), 1e-12f);
      }
    }
  }
  #pragma unroll
  for (int c = 0; c < NCAPS; ++c)
    out[(size_t)node * 256 + c * CAPD + lane] = u[c];
}

// ------------------------------- GCN agg ----------------------------------

__global__ void fill1_kernel(float* __restrict__ deg, int n) {
  int i = blockIdx.x * blockDim.x + threadIdx.x;
  if (i < n) deg[i] = 1.0f;                 // self-loop contribution
}

__global__ void deg_kernel(const int* __restrict__ col, int E, float* __restrict__ deg) {
  for (int e = blockIdx.x * blockDim.x + threadIdx.x; e < E; e += gridDim.x * blockDim.x)
    atomicAdd(&deg[col[e]], 1.0f);
}

__global__ void dinv_kernel(const float* __restrict__ deg, float* __restrict__ dinv, int n) {
  int i = blockIdx.x * blockDim.x + threadIdx.x;
  if (i < n) dinv[i] = rsqrtf(deg[i]);
}

__global__ void outinit_kernel(const float* __restrict__ r, const float* __restrict__ dinv,
                               float* __restrict__ out, int total) {
  int idx = blockIdx.x * blockDim.x + threadIdx.x;
  if (idx >= total) return;
  int node = idx >> 8;
  float d = dinv[node];
  out[idx] = r[idx] * d * d;                // self-loop term
}

// one wave per edge: out[row] += r[col] * dinv[row]*dinv[col]
__global__ void spmm_kernel(const int* __restrict__ row, const int* __restrict__ col,
                            const float* __restrict__ r, const float* __restrict__ dinv,
                            float* __restrict__ out, int E) {
  int e    = (blockIdx.x * blockDim.x + threadIdx.x) >> 5;
  int lane = threadIdx.x & 31;
  if (e >= E) return;
  int rr = row[e], cc = col[e];
  float w = dinv[rr] * dinv[cc];
  const float* src = r   + (size_t)cc * 256;
  float*       dst = out + (size_t)rr * 256;
  #pragma unroll
  for (int t = 0; t < 8; ++t)
    atomicAdd(&dst[t * 32 + lane], src[t * 32 + lane] * w);
}

// ------------------------------- launcher ---------------------------------

extern "C" void kernel_launch(void* const* d_in, const int* in_sizes, int n_in,
                              void* d_out, int out_size, void* d_ws, size_t ws_size,
                              hipStream_t stream) {
  const float* x      = (const float*)d_in[0];
  const float* W1     = (const float*)d_in[1];
  const float* b1     = (const float*)d_in[2];
  const float* gamma1 = (const float*)d_in[3];
  const float* beta1  = (const float*)d_in[4];
  const float* W2     = (const float*)d_in[5];
  const float* b2     = (const float*)d_in[6];
  const float* gamma2 = (const float*)d_in[7];
  const float* beta2  = (const float*)d_in[8];
  const int*   edges  = (const int*)d_in[9];
  const int*   nbr    = (const int*)d_in[10];

  const int NN = in_sizes[0] / NODES_IN;   // 50000
  const int E  = in_sizes[9] / 2;          // 1,000,000

  char* ws = (char*)d_ws;
  size_t off = 0;
  auto alloc = [&](size_t bytes) -> char* {
    char* p = ws + off;
    off += (bytes + 255) & ~(size_t)255;
    return p;
  };
  __bf16* xb   = (__bf16*)alloc((size_t)NN * KPAD * 2);   // reused as h1b later
  float*  h1   = (float*) alloc((size_t)NN * HID * 4);    // reused as routing bufB
  float*  bufA = (float*) alloc((size_t)NN * OUTD * 4);   // h2 / routing ping
  __bf16* w1t  = (__bf16*)alloc((size_t)HID  * KPAD * 2);
  __bf16* w2t  = (__bf16*)alloc((size_t)OUTD * KPAD * 2);
  float* mean1 = (float*)alloc(HID * 4);
  float* rstd1 = (float*)alloc(HID * 4);
  float* mean2 = (float*)alloc(OUTD * 4);
  float* rstd2 = (float*)alloc(OUTD * 4);
  float* deg   = (float*)alloc((size_t)NN * 4);
  float* dinv  = (float*)alloc((size_t)NN * 4);
  __bf16* h1b  = xb;    // alias: xb dead after GEMM1
  float*  bufB = h1;    // alias: h1 dead after bnapply_bf16

  const int tM = NN / 16;                  // 3125 row tiles

  // --- MLP trunk ---
  conv_x_kernel <<<(NN * KPAD + 255) / 256, 256, 0, stream>>>(x, xb, NN);
  conv_w1_kernel<<<(HID  * KPAD) / 256, 256, 0, stream>>>(W1, w1t);
  conv_w2_kernel<<<(OUTD * KPAD) / 256, 256, 0, stream>>>(W2, w2t);

  gemm_bf16_kernel<<<dim3(HID / 256, tM), 256, 0, stream>>>(xb, w1t, b1, h1, HID);
  bnstats_kernel  <<<HID, 256, 0, stream>>>(h1, HID, NN, mean1, rstd1);
  bnapply_bf16_kernel<<<(NN * HID + 255) / 256, 256, 0, stream>>>(
      h1, mean1, rstd1, gamma1, beta1, h1b, NN * HID, HID - 1);

  gemm_bf16_kernel<<<dim3(OUTD / 256, tM), 256, 0, stream>>>(h1b, w2t, b2, bufA, OUTD);
  bnstats_kernel  <<<OUTD, 256, 0, stream>>>(bufA, OUTD, NN, mean2, rstd2);
  bnapply_f32_kernel<<<(NN * OUTD + 255) / 256, 256, 0, stream>>>(
      bufA, mean2, rstd2, gamma2, beta2, bufA, NN * OUTD, OUTD - 1);

  // --- capsule routing (nlayer = 2, max_iter = 3, k = 8) ---
  const size_t route_lds = 2 * NNBR * 256 * sizeof(float);   // 40 KB
  for (int layer = 0; layer < 2; ++layer) {
    capsnorm_kernel<<<(NN + 7) / 8, 256, 0, stream>>>(bufA, bufB, NN);
    route_kernel   <<<(NN + 1) / 2, 64, route_lds, stream>>>(bufB, nbr, bufA, NN);
  }

  // --- GCN aggregation ---
  fill1_kernel<<<(NN + 255) / 256, 256, 0, stream>>>(deg, NN);
  deg_kernel  <<<4096, 256, 0, stream>>>(edges + E, E, deg);   // col = edges[1]
  dinv_kernel <<<(NN + 255) / 256, 256, 0, stream>>>(deg, dinv, NN);
  outinit_kernel<<<(NN * OUTD + 255) / 256, 256, 0, stream>>>(
      bufA, dinv, (float*)d_out, NN * OUTD);
  spmm_kernel<<<((E * 32) + 255) / 256, 256, 0, stream>>>(
      edges, edges + E, bufA, dinv, (float*)d_out, E);
}
